// SimpleAttention_91001767068477
// MI455X (gfx1250) — matrix operand order repaired and checked by reading, not compiled, not run on hardware
//
#include <hip/hip_runtime.h>

// Problem constants (from reference): B=16, C=512, H=W=64 -> N=4096
#define BATCH 16
#define CDIM  512
#define NDIM  4096

typedef __attribute__((ext_vector_type(16))) __bf16 v16bf;
typedef __attribute__((ext_vector_type(8)))  __bf16 v8bf;
typedef __attribute__((ext_vector_type(8)))  float  v8f;
typedef __attribute__((ext_vector_type(4)))  int    v4i;

// Tiling: block tile 128x128, K-step 32 (one bf16 WMMA K depth).
// 256 threads = 8 wave32; wave grid 2(M) x 4(N); each wave: 64x32 = 4x2 frags.
#define BM 128
#define BN 128
#define BK 32
#define LDT 40   // padded LDS row stride in bf16 (80B rows)

// ---------------------------------------------------------------------------
// CDNA5 async global->LDS copy (ASYNCcnt-tracked). Signature (from clang
// diagnostic): param0 = addrspace(1) v4i*, then LDS ptr, offset, cpol.
// Guarded so a toolchain without the builtins falls back to sync copies.
// ---------------------------------------------------------------------------
#if __has_builtin(__builtin_amdgcn_global_load_async_to_lds_b128) && \
    __has_builtin(__builtin_amdgcn_s_wait_asynccnt)
#define USE_ASYNC_LDS 1
#else
#define USE_ASYNC_LDS 0
#endif

__device__ __forceinline__ void copy_b128_tile(__bf16* lds, const __bf16* g) {
#if USE_ASYNC_LDS
  __builtin_amdgcn_global_load_async_to_lds_b128(
      (__attribute__((address_space(1))) v4i*)(uintptr_t)g,
      (__attribute__((address_space(3))) v4i*)(uint32_t)(uintptr_t)lds,
      /*offset=*/0, /*cpol=*/0);
#else
  *(v8bf*)lds = *(const v8bf*)g;
#endif
}

#if USE_ASYNC_LDS
#define WAIT_ASYNC(lit) __builtin_amdgcn_s_wait_asynccnt(lit)
#else
#define WAIT_ASYNC(lit) ((void)0)
#endif

__device__ __forceinline__ v8f zero_v8f() {
  v8f z;
#pragma unroll
  for (int e = 0; e < 8; ++e) z[e] = 0.0f;
  return z;
}

// A fragment (16x32 bf16, MxK). Lane L<16: M=L, K in {0..7, 16..23};
// lane L>=16: M=L-16, K in {8..15, 24..31}.
__device__ __forceinline__ v16bf load_a_frag(const __bf16* As, int row, int lane) {
  const __bf16* p = As + row * LDT + ((lane & 16) ? 8 : 0);
  v8bf a0 = *(const v8bf*)(p);
  v8bf a1 = *(const v8bf*)(p + 16);
  v16bf a;
#pragma unroll
  for (int i = 0; i < 8; ++i) { a[i] = a0[i]; a[8 + i] = a1[i]; }
  return a;
}

// B fragment (32x16 bf16, KxN), LDS stored as [N][K]. Lane L<16: N=L,
// K in {0..7, 16..23}; lane L>=16: N=L-16, K in {8..15, 24..31}.
__device__ __forceinline__ v16bf load_b_frag(const __bf16* Bs, int col, int lane) {
  const __bf16* p = Bs + col * LDT + ((lane & 16) ? 8 : 0);
  v8bf b0 = *(const v8bf*)(p);
  v8bf b1 = *(const v8bf*)(p + 16);
  v16bf b;
#pragma unroll
  for (int i = 0; i < 8; ++i) { b[i] = b0[i]; b[8 + i] = b1[i]; }
  return b;
}

// ---------------------------------------------------------------------------
// Kernel 1: q/k/v = W @ x + b   (per batch; W fp32 [C,C], x fp32 [C,N])
// q,k stored bf16 [B,C,N]; v stored TRANSPOSED bf16 [B,N,C] so that the
// final GEMM's B-tile becomes a contiguous (async-able) row copy.
// gridDim.z = BATCH*3 -> (batch, which-of-qkv).
// ---------------------------------------------------------------------------
__global__ __launch_bounds__(256) void qkv_gemm(
    const float* __restrict__ x,
    const float* __restrict__ Wq, const float* __restrict__ bq,
    const float* __restrict__ Wk, const float* __restrict__ bk,
    const float* __restrict__ Wv, const float* __restrict__ bv,
    __bf16* __restrict__ qws, __bf16* __restrict__ kws, __bf16* __restrict__ vtws)
{
  __shared__ __bf16 As[BM * LDT];
  __shared__ __bf16 Bs[BN * LDT];

  const int tid  = threadIdx.x;
  const int lane = tid & 31;
  const int wave = tid >> 5;
  const int lm   = lane & 15;
  const int hi8  = (lane & 16) ? 8 : 0;

  const int nBase = blockIdx.x * BN;
  const int mBase = blockIdx.y * BM;
  const int z  = blockIdx.z;
  const int bb = z / 3;
  const int which = z - bb * 3;

  const float* Wmat = (which == 0) ? Wq : (which == 1) ? Wk : Wv;
  const float* bias = (which == 0) ? bq : (which == 1) ? bk : bv;
  const float* xb = x + (size_t)bb * CDIM * NDIM;

  const int wmOff = (wave & 1) * 64;
  const int wnOff = (wave >> 1) * 32;

  v8f acc[4][2];
#pragma unroll
  for (int i = 0; i < 4; ++i)
#pragma unroll
    for (int j = 0; j < 2; ++j) acc[i][j] = zero_v8f();

  for (int k0 = 0; k0 < CDIM; k0 += BK) {
    // Prefetch next K-tile sources into cache while converting this one.
    if (k0 + BK < CDIM) {
      __builtin_prefetch(Wmat + (size_t)(mBase + (tid >> 3)) * CDIM + k0 + BK, 0, 3);
      __builtin_prefetch(xb + (size_t)(k0 + BK + (tid >> 5)) * NDIM + nBase + ((tid & 31) << 2), 0, 3);
    }
    // A tile: W[mBase+r][k0+c], fp32 -> bf16, row-major [M][K]
    {
      const int r0 = tid >> 3;
      const int c4 = (tid & 7) << 2;
#pragma unroll
      for (int p = 0; p < 4; ++p) {
        const int r = r0 + p * 32;
        const float4 f = *(const float4*)(Wmat + (size_t)(mBase + r) * CDIM + k0 + c4);
        __bf16* d = &As[r * LDT + c4];
        d[0] = (__bf16)f.x; d[1] = (__bf16)f.y; d[2] = (__bf16)f.z; d[3] = (__bf16)f.w;
      }
    }
    // B tile transposed: Bs[n][c] = x[k0+c][nBase+n], fp32 -> bf16
    {
      const int cc = tid >> 5;
      const int n4 = (tid & 31) << 2;
#pragma unroll
      for (int p = 0; p < 4; ++p) {
        const int c = cc + p * 8;
        const float4 f = *(const float4*)(xb + (size_t)(k0 + c) * NDIM + nBase + n4);
        Bs[(n4 + 0) * LDT + c] = (__bf16)f.x;
        Bs[(n4 + 1) * LDT + c] = (__bf16)f.y;
        Bs[(n4 + 2) * LDT + c] = (__bf16)f.z;
        Bs[(n4 + 3) * LDT + c] = (__bf16)f.w;
      }
    }
    __syncthreads();

    v16bf af[4];
#pragma unroll
    for (int i = 0; i < 4; ++i) af[i] = load_a_frag(As, wmOff + i * 16 + lm, lane);
    v16bf bfr[2];
#pragma unroll
    for (int j = 0; j < 2; ++j) bfr[j] = load_b_frag(Bs, wnOff + j * 16 + lm, lane);
#pragma unroll
    for (int i = 0; i < 4; ++i)
#pragma unroll
      for (int j = 0; j < 2; ++j)
        acc[i][j] = __builtin_amdgcn_wmma_f32_16x16x32_bf16(
            false, af[i], false, bfr[j], (short)0, acc[i][j], false, false);
    __syncthreads();
  }

  // Epilogue: + bias; q,k row-major, v transposed.
  if (which == 2) {
    __bf16* vt = vtws + (size_t)bb * NDIM * CDIM;
#pragma unroll
    for (int i = 0; i < 4; ++i)
#pragma unroll
      for (int j = 0; j < 2; ++j)
#pragma unroll
        for (int r = 0; r < 8; ++r) {
          const int row = mBase + wmOff + i * 16 + hi8 + r;   // c
          const int col = nBase + wnOff + j * 16 + lm;        // n
          vt[(size_t)col * CDIM + row] = (__bf16)(acc[i][j][r] + bias[row]);
        }
  } else {
    __bf16* outp = ((which == 0) ? qws : kws) + (size_t)bb * CDIM * NDIM;
#pragma unroll
    for (int i = 0; i < 4; ++i)
#pragma unroll
      for (int j = 0; j < 2; ++j)
#pragma unroll
        for (int r = 0; r < 8; ++r) {
          const int row = mBase + wmOff + i * 16 + hi8 + r;
          const int col = nBase + wnOff + j * 16 + lm;
          outp[(size_t)row * NDIM + col] = (__bf16)(acc[i][j][r] + bias[row]);
        }
  }
}

// ---------------------------------------------------------------------------
// Kernel 2: S[b] = (q[b] @ k[b]^T) / sqrt(C); K = n contiguous for both.
// Async double-buffered LDS tiles (4 async b128 ops per thread per tile pair).
// ---------------------------------------------------------------------------
__global__ __launch_bounds__(256) void attn_gemm(
    const __bf16* __restrict__ qws, const __bf16* __restrict__ kws,
    float* __restrict__ Sws)
{
  __shared__ __bf16 As[2][BM * LDT];
  __shared__ __bf16 Bs[2][BN * LDT];

  const int tid  = threadIdx.x;
  const int lane = tid & 31;
  const int wave = tid >> 5;
  const int lm   = lane & 15;
  const int hi8  = (lane & 16) ? 8 : 0;

  const int dBase = blockIdx.x * BN;
  const int cBase = blockIdx.y * BM;
  const int bb    = blockIdx.z;

  const __bf16* qb = qws + (size_t)bb * CDIM * NDIM;
  const __bf16* kb = kws + (size_t)bb * CDIM * NDIM;
  float* Sb = Sws + (size_t)bb * CDIM * CDIM;

  const int wmOff = (wave & 1) * 64;
  const int wnOff = (wave >> 1) * 32;

  v8f acc[4][2];
#pragma unroll
  for (int i = 0; i < 4; ++i)
#pragma unroll
    for (int j = 0; j < 2; ++j) acc[i][j] = zero_v8f();

  auto issue_tiles = [&](int k0, int buf) {
#pragma unroll
    for (int p = 0; p < 2; ++p) {
      const int vI = tid + p * 256;
      const int r  = vI >> 2;
      const int c8 = (vI & 3) << 3;
      copy_b128_tile(&As[buf][r * LDT + c8], qb + (size_t)(cBase + r) * NDIM + k0 + c8);
      copy_b128_tile(&Bs[buf][r * LDT + c8], kb + (size_t)(dBase + r) * NDIM + k0 + c8);
    }
  };

  constexpr int KT = NDIM / BK;
  issue_tiles(0, 0);
  int cur = 0;
  for (int kt = 0; kt < KT; ++kt) {
    if (kt + 1 < KT) {
      issue_tiles((kt + 1) * BK, cur ^ 1);
      WAIT_ASYNC(4);   // in-order: first 4 (current tile) retired
    } else {
      WAIT_ASYNC(0);
    }
    __syncthreads();

    v16bf af[4];
#pragma unroll
    for (int i = 0; i < 4; ++i) af[i] = load_a_frag(As[cur], wmOff + i * 16 + lm, lane);
    v16bf bfr[2];
#pragma unroll
    for (int j = 0; j < 2; ++j) bfr[j] = load_b_frag(Bs[cur], wnOff + j * 16 + lm, lane);
#pragma unroll
    for (int i = 0; i < 4; ++i)
#pragma unroll
      for (int j = 0; j < 2; ++j)
        acc[i][j] = __builtin_amdgcn_wmma_f32_16x16x32_bf16(
            false, af[i], false, bfr[j], (short)0, acc[i][j], false, false);
    __syncthreads();
    cur ^= 1;
  }

  const float scale = 0.044194173824159216f;  // 1/sqrt(512)
#pragma unroll
  for (int i = 0; i < 4; ++i)
#pragma unroll
    for (int j = 0; j < 2; ++j)
#pragma unroll
      for (int r = 0; r < 8; ++r) {
        const int row = cBase + wmOff + i * 16 + hi8 + r;
        const int col = dBase + wnOff + j * 16 + lm;
        Sb[(size_t)row * CDIM + col] = acc[i][j][r] * scale;
      }
}

// ---------------------------------------------------------------------------
// Kernel 3: row softmax over 512-wide rows of S; output bf16 attn
// ---------------------------------------------------------------------------
__global__ __launch_bounds__(256) void softmax_rows(
    const float* __restrict__ S, __bf16* __restrict__ A)
{
  __shared__ float red[256];
  const int row = blockIdx.x;
  const int t = threadIdx.x;
  const float* srow = S + (size_t)row * CDIM;

  const float v0 = srow[t];
  const float v1 = srow[t + 256];

  red[t] = fmaxf(v0, v1);
  __syncthreads();
#pragma unroll
  for (int s = 128; s > 0; s >>= 1) {
    if (t < s) red[t] = fmaxf(red[t], red[t + s]);
    __syncthreads();
  }
  const float m = red[0];
  __syncthreads();

  const float e0 = __expf(v0 - m);
  const float e1 = __expf(v1 - m);
  red[t] = e0 + e1;
  __syncthreads();
#pragma unroll
  for (int s = 128; s > 0; s >>= 1) {
    if (t < s) red[t] += red[t + s];
    __syncthreads();
  }
  const float inv = 1.0f / red[0];

  __bf16* arow = A + (size_t)row * CDIM;
  arow[t]       = (__bf16)(e0 * inv);
  arow[t + 256] = (__bf16)(e1 * inv);
}

// ---------------------------------------------------------------------------
// Kernel 4: out[b] = attn[b] @ v[b].  attn bf16 [C,C]; v stored transposed
// [N,C], so the B tile Bs[n][d] is a contiguous row copy. Async + double
// buffered like attn_gemm.
// ---------------------------------------------------------------------------
__global__ __launch_bounds__(256) void out_gemm(
    const __bf16* __restrict__ Aws, const __bf16* __restrict__ vtws,
    float* __restrict__ out)
{
  __shared__ __bf16 As[2][BM * LDT];
  __shared__ __bf16 Bs[2][BN * LDT];

  const int tid  = threadIdx.x;
  const int lane = tid & 31;
  const int wave = tid >> 5;
  const int lm   = lane & 15;
  const int hi8  = (lane & 16) ? 8 : 0;

  const int nBase = blockIdx.x * BN;
  const int cBase = blockIdx.y * BM;
  const int bb    = blockIdx.z;

  const __bf16* ab = Aws  + (size_t)bb * CDIM * CDIM;
  const __bf16* vt = vtws + (size_t)bb * NDIM * CDIM;
  float* ob = out + (size_t)bb * CDIM * NDIM;

  const int wmOff = (wave & 1) * 64;
  const int wnOff = (wave >> 1) * 32;

  v8f acc[4][2];
#pragma unroll
  for (int i = 0; i < 4; ++i)
#pragma unroll
    for (int j = 0; j < 2; ++j) acc[i][j] = zero_v8f();

  auto issue_tiles = [&](int k0, int buf) {
#pragma unroll
    for (int p = 0; p < 2; ++p) {
      const int vI = tid + p * 256;
      const int r  = vI >> 2;
      const int c8 = (vI & 3) << 3;
      // A: attn[cBase+r][k0+c8..]  (K = d contiguous)
      copy_b128_tile(&As[buf][r * LDT + c8], ab + (size_t)(cBase + r) * CDIM + k0 + c8);
      // B: vT[nBase+r][k0+c8..]    (K = d contiguous)
      copy_b128_tile(&Bs[buf][r * LDT + c8], vt + (size_t)(nBase + r) * CDIM + k0 + c8);
    }
  };

  constexpr int KT = CDIM / BK;
  issue_tiles(0, 0);
  int cur = 0;
  for (int kt = 0; kt < KT; ++kt) {
    if (kt + 1 < KT) {
      issue_tiles((kt + 1) * BK, cur ^ 1);
      WAIT_ASYNC(4);
    } else {
      WAIT_ASYNC(0);
    }
    __syncthreads();

    v16bf af[4];
#pragma unroll
    for (int i = 0; i < 4; ++i) af[i] = load_a_frag(As[cur], wmOff + i * 16 + lm, lane);
    v16bf bfr[2];
#pragma unroll
    for (int j = 0; j < 2; ++j) bfr[j] = load_b_frag(Bs[cur], wnOff + j * 16 + lm, lane);
#pragma unroll
    for (int i = 0; i < 4; ++i)
#pragma unroll
      for (int j = 0; j < 2; ++j)
        acc[i][j] = __builtin_amdgcn_wmma_f32_16x16x32_bf16(
            false, af[i], false, bfr[j], (short)0, acc[i][j], false, false);
    __syncthreads();
    cur ^= 1;
  }

#pragma unroll
  for (int i = 0; i < 4; ++i)
#pragma unroll
    for (int j = 0; j < 2; ++j)
#pragma unroll
      for (int r = 0; r < 8; ++r) {
        const int row = cBase + wmOff + i * 16 + hi8 + r;
        const int col = nBase + wnOff + j * 16 + lm;
        ob[(size_t)row * NDIM + col] = acc[i][j][r];
      }
}

// ---------------------------------------------------------------------------
// Host launcher
// ---------------------------------------------------------------------------
extern "C" void kernel_launch(void* const* d_in, const int* in_sizes, int n_in,
                              void* d_out, int out_size, void* d_ws, size_t ws_size,
                              hipStream_t stream) {
  (void)in_sizes; (void)n_in; (void)out_size; (void)ws_size;

  const float* x  = (const float*)d_in[0];
  const float* Wq = (const float*)d_in[1];
  const float* bq = (const float*)d_in[2];
  const float* Wk = (const float*)d_in[3];
  const float* bk = (const float*)d_in[4];
  const float* Wv = (const float*)d_in[5];
  const float* bv = (const float*)d_in[6];
  float* out = (float*)d_out;

  // Workspace layout:
  //   q  bf16 [B,C,N]  (64 MiB)
  //   k  bf16 [B,C,N]  (64 MiB)
  //   vT bf16 [B,N,C]  (64 MiB)
  //   S  f32  [B,C,C]  (16 MiB)
  //   A  bf16 [B,C,C]  ( 8 MiB)
  const size_t qkvElems = (size_t)BATCH * CDIM * NDIM;
  char* base = (char*)d_ws;
  __bf16* qws  = (__bf16*)(base);
  __bf16* kws  = (__bf16*)(base + qkvElems * 2);
  __bf16* vtws = (__bf16*)(base + qkvElems * 4);
  float*  Sws  = (float*) (base + qkvElems * 6);
  __bf16* Aws  = (__bf16*)(base + qkvElems * 6 + (size_t)BATCH * CDIM * CDIM * 4);

  dim3 blk(256, 1, 1);

  dim3 g1(NDIM / BN, CDIM / BM, BATCH * 3);
  qkv_gemm<<<g1, blk, 0, stream>>>(x, Wq, bq, Wk, bk, Wv, bv, qws, kws, vtws);

  dim3 g2(CDIM / BN, CDIM / BM, BATCH);
  attn_gemm<<<g2, blk, 0, stream>>>(qws, kws, Sws);

  dim3 g3(BATCH * CDIM, 1, 1);
  softmax_rows<<<g3, blk, 0, stream>>>(Sws, Aws);

  dim3 g4(NDIM / BN, CDIM / BM, BATCH);
  out_gemm<<<g4, blk, 0, stream>>>(Aws, vtws, out);
}
